// TargetAttention_76201309766101
// MI455X (gfx1250) — matrix operand order repaired and checked by reading, not compiled, not run on hardware
//
#include <hip/hip_runtime.h>
#include <hip/hip_bf16.h>
#include <math.h>
#include <stdint.h>

typedef __attribute__((ext_vector_type(16))) _Float16 v16h;
typedef __attribute__((ext_vector_type(8)))  _Float16 v8h;
typedef __attribute__((ext_vector_type(8)))  float    v8f;

#define D_    64
#define T_    200
#define H_    128
#define TP    208    // T padded to 13 tiles of 16
#define HSTR  72     // shist row stride in halves (144B: bank-decorrelating, 16B aligned)
#define H1STR 136    // h1buf row stride in halves (272B)

#define NB1   (8 * 8 * 32 * 16)   // 32768 W1 fragment halves (64 KB)
#define NB2   (4 * 4 * 32 * 16)   // 8192  W2 fragment halves (16 KB)

__device__ __forceinline__ float gelu_f(float x) {
    // exact gelu: x * 0.5 * (1 + erf(x / sqrt(2)))
    return 0.5f * x * (1.0f + erff(x * 0.70710678f));
}

// ---- one-time weight swizzle: W1/W2 (f32 row-major) -> f16 WMMA B-fragment layout ----
// B-fragment: [ks][nt][lane][e], col = lane&15, K = e + 16*(lane>=16)
__global__ __launch_bounds__(256)
void prep_weights_kernel(const float* __restrict__ W1, const float* __restrict__ W2,
                         _Float16* __restrict__ wf) {
    int i = blockIdx.x * 256 + threadIdx.x;
    if (i < NB1) {
        int e = i & 15, ln = (i >> 4) & 31, nt = (i >> 9) & 7, ks = i >> 12;
        int k = ks * 32 + e + ((ln >> 4) << 4);
        int n = nt * 16 + (ln & 15);
        wf[i] = (_Float16)W1[k * H_ + n];
    } else if (i < NB1 + NB2) {
        int j = i - NB1;
        int e = j & 15, ln = (j >> 4) & 31, nt = (j >> 9) & 3, ks = j >> 11;
        int k = ks * 32 + e + ((ln >> 4) << 4);
        int n = nt * 16 + (ln & 15);
        wf[i] = (_Float16)W2[k * D_ + n];
    }
}

template <int USE_WS>
__global__ __launch_bounds__(256, 1)
void ta_mlp_attn_kernel(const float* __restrict__ hist,
                        const float* __restrict__ targ,
                        const int*   __restrict__ mask,
                        const float* __restrict__ W1, const float* __restrict__ b1,
                        const float* __restrict__ W2, const float* __restrict__ b2,
                        const float* __restrict__ W3, const float* __restrict__ b3,
                        const _Float16* __restrict__ wfrag,
                        float* __restrict__ out)
{
    __shared__ __align__(16) _Float16 shist[TP][HSTR];          // ~29.9 KB
    __shared__ __align__(16) float    sq[D_];
    __shared__ __align__(32) _Float16 sB1[8][8][32][16];        // 64 KB  W1 fragments
    __shared__ __align__(32) _Float16 sB2[4][4][32][16];        // 16 KB  W2 fragments
    __shared__ __align__(16) _Float16 h1buf[8][16][H1STR];      // ~34.8 KB wave-private
    __shared__ float sb1[H_];
    __shared__ float sb2[D_];
    __shared__ float sw3[D_];
    __shared__ float sb3;
    __shared__ float sScore[TP];
    __shared__ float sWgt[TP];
    __shared__ float sRed[256];

    const int tid    = threadIdx.x;
    const int b      = blockIdx.x;
    const int lane   = tid & 31;
    const int wave   = tid >> 5;
    const int lrow   = lane & 15;
    const int half   = lane >> 4;      // 0 or 1
    const int khalf  = half << 4;      // A-fragment K offset (0 / 16)
    const int rowoff = half << 3;      // C/D row offset      (0 / 8)

    // ---------------- weight fragments -> LDS ----------------
    if (USE_WS) {
        // Async DMA of pre-swizzled fragments straight into LDS (no VGPR bounce),
        // overlapped with the history conversion below; tracked by ASYNCcnt.
        const uint32_t ldsB1 = (uint32_t)(uintptr_t)&sB1[0][0][0][0];
        const uint32_t ldsB2 = (uint32_t)(uintptr_t)&sB2[0][0][0][0];
        const uint64_t gbase = (uint64_t)(uintptr_t)wfrag;
        #pragma unroll
        for (int c = 0; c < 16; c++) {                 // 16 x 256 lanes x 16B = 64 KB
            uint32_t byteoff = (uint32_t)(c * 256 + tid) * 16u;
            uint32_t la = ldsB1 + byteoff;
            asm volatile("global_load_async_to_lds_b128 %0, %1, %2 offset:0"
                         :: "v"(la), "v"(byteoff), "s"(gbase) : "memory");
        }
        #pragma unroll
        for (int c = 0; c < 4; c++) {                  // 16 KB
            uint32_t byteoff = (uint32_t)(c * 256 + tid) * 16u;
            uint32_t la = ldsB2 + byteoff;
            uint32_t go = (uint32_t)(NB1 * 2) + byteoff;
            asm volatile("global_load_async_to_lds_b128 %0, %1, %2 offset:0"
                         :: "v"(la), "v"(go), "s"(gbase) : "memory");
        }
    } else {
        // Fallback: in-kernel scatter conversion (ws too small).
        for (int i = tid; i < NB1; i += 256) {
            int e = i & 15, ln = (i >> 4) & 31, nt = (i >> 9) & 7, ks = i >> 12;
            int k = ks * 32 + e + ((ln >> 4) << 4);
            int n = nt * 16 + (ln & 15);
            sB1[ks][nt][ln][e] = (_Float16)W1[k * H_ + n];
        }
        for (int i = tid; i < NB2; i += 256) {
            int e = i & 15, ln = (i >> 4) & 31, nt = (i >> 9) & 3, ks = i >> 11;
            int k = ks * 32 + e + ((ln >> 4) << 4);
            int n = nt * 16 + (ln & 15);
            sB2[ks][nt][ln][e] = (_Float16)W2[k * D_ + n];
        }
    }

    // ---------------- stage history / q / biases ----------------
    const float* hsrc = hist + (size_t)b * (T_ * D_);
    for (int i = tid; i < T_ * D_; i += 256)
        shist[i >> 6][i & 63] = (_Float16)hsrc[i];
    for (int i = tid; i < (TP - T_) * HSTR; i += 256)
        shist[T_ + i / HSTR][i % HSTR] = (_Float16)0.f;
    if (tid < D_) sq[tid] = targ[(size_t)b * D_ + tid];
    if (tid < H_) sb1[tid] = b1[tid];
    if (tid < D_) { sb2[tid] = b2[tid]; sw3[tid] = W3[tid]; }
    if (tid == 0) sb3 = b3[0];

    if (USE_WS)
        asm volatile("s_wait_asynccnt 0x0" ::: "memory");
    __syncthreads();

    // ---------------- per-M-tile MLP (each wave owns tiles wave, wave+8) ----------------
    for (int mt = wave; mt < 13; mt += 8) {
        const int trow = mt * 16 + lrow;   // token row this lane supplies to A-fragments

        // ---- GEMM1: feat(16x256) @ W1(256x128), f16 in / f32 acc ----
        v8f acc[8];
        #pragma unroll
        for (int nt = 0; nt < 8; nt++)
            #pragma unroll
            for (int r = 0; r < 8; r++) acc[nt][r] = 0.f;

        #pragma unroll
        for (int ks = 0; ks < 8; ks++) {
            const int quad  = ks >> 1;                    // 0:h 1:q 2:h*q 3:|h-q|
            const int cbase = ((ks & 1) << 5) + khalf;    // column base within D=64
            v8h h0  = *(const v8h*)&shist[trow][cbase];
            v8h h1v = *(const v8h*)&shist[trow][cbase + 8];
            v16h a;
            #pragma unroll
            for (int e = 0; e < 8; e++) {
                float hv = (float)h0[e],  qv = sq[cbase + e];
                float hw = (float)h1v[e], qw = sq[cbase + 8 + e];
                float f0, f1;
                if (quad == 0)      { f0 = hv;             f1 = hw; }
                else if (quad == 1) { f0 = qv;             f1 = qw; }
                else if (quad == 2) { f0 = hv * qv;        f1 = hw * qw; }
                else {
                    // |h - q| without the fabs idiom: SLP pairing of the two
                    // fsub+fabs chains into v2f32 crashes gfx1250 ISel.
                    float d0 = hv - qv;
                    float d1 = hw - qw;
                    asm("" : "+v"(d0));   // opaque copies: keep these scalar
                    asm("" : "+v"(d1));
                    f0 = fmaxf(d0, -d0);  // == |d0| (finite inputs)
                    f1 = fmaxf(d1, -d1);
                }
                a[e]     = (_Float16)f0;
                a[e + 8] = (_Float16)f1;
            }
            #pragma unroll
            for (int nt = 0; nt < 8; nt++) {
                v16h bf = *(const v16h*)&sB1[ks][nt][lane][0];
                acc[nt] = __builtin_amdgcn_wmma_f32_16x16x32_f16(
                    false, a, false, bf, (short)0, acc[nt], false, false);
            }
        }

        // ---- bias + gelu -> h1buf (f16, row-major per wave) ----
        #pragma unroll
        for (int nt = 0; nt < 8; nt++) {
            const int col  = nt * 16 + lrow;
            const float bs = sb1[col];
            #pragma unroll
            for (int r = 0; r < 8; r++) {
                float v = gelu_f(acc[nt][r] + bs);
                h1buf[wave][r + rowoff][col] = (_Float16)v;
            }
        }
        // DS ops from the same wave are in-order: reads below see the stores above.

        // ---- GEMM2: h1(16x128) @ W2(128x64) ----
        v8f acc2[4];
        #pragma unroll
        for (int nt = 0; nt < 4; nt++)
            #pragma unroll
            for (int r = 0; r < 8; r++) acc2[nt][r] = 0.f;

        #pragma unroll
        for (int ks = 0; ks < 4; ks++) {
            const _Float16* hb = &h1buf[wave][lrow][0];
            v8h a0 = *(const v8h*)&hb[ks * 32 + khalf];
            v8h a1 = *(const v8h*)&hb[ks * 32 + khalf + 8];
            v16h a;
            #pragma unroll
            for (int e = 0; e < 8; e++) { a[e] = a0[e]; a[e + 8] = a1[e]; }
            #pragma unroll
            for (int nt = 0; nt < 4; nt++) {
                v16h bf = *(const v16h*)&sB2[ks][nt][lane][0];
                acc2[nt] = __builtin_amdgcn_wmma_f32_16x16x32_f16(
                    false, a, false, bf, (short)0, acc2[nt], false, false);
            }
        }

        // ---- layer3: gelu(h2)·W3 -> per-token score ----
        float p[8];
        #pragma unroll
        for (int r = 0; r < 8; r++) p[r] = 0.f;
        #pragma unroll
        for (int nt = 0; nt < 4; nt++) {
            const int col  = nt * 16 + lrow;
            const float bs = sb2[col];
            const float w3 = sw3[col];
            #pragma unroll
            for (int r = 0; r < 8; r++)
                p[r] += gelu_f(acc2[nt][r] + bs) * w3;
        }
        // reduce across the 16 lanes of each half-wave (rows differ between halves)
        #pragma unroll
        for (int r = 0; r < 8; r++) {
            p[r] += __shfl_xor(p[r], 1, 16);
            p[r] += __shfl_xor(p[r], 2, 16);
            p[r] += __shfl_xor(p[r], 4, 16);
            p[r] += __shfl_xor(p[r], 8, 16);
        }
        if (lrow == 0) {
            const float bias3 = sb3;
            #pragma unroll
            for (int r = 0; r < 8; r++)
                sScore[mt * 16 + r + rowoff] = gelu_f(p[r] + bias3);
        }
    }
    __syncthreads();

    // ---------------- masked softmax over T ----------------
    float s = -INFINITY;
    if (tid < T_) {
        s = sScore[tid];
        if (mask[(size_t)b * T_ + tid] != 0) s = -INFINITY;
    }
    sRed[tid] = s;
    __syncthreads();
    for (int o = 128; o > 0; o >>= 1) {
        if (tid < o) sRed[tid] = fmaxf(sRed[tid], sRed[tid + o]);
        __syncthreads();
    }
    const float mx = sRed[0];
    __syncthreads();

    float w = (tid < T_) ? expf(s - mx) : 0.f;
    sRed[tid] = w;
    __syncthreads();
    for (int o = 128; o > 0; o >>= 1) {
        if (tid < o) sRed[tid] = sRed[tid] + sRed[tid + o];
        __syncthreads();
    }
    const float total = sRed[0];
    if (tid < TP) sWgt[tid] = (tid < T_) ? (w / total) : 0.f;
    __syncthreads();

    // ---------------- out[b] = sum_t weight[t] * hist[b,t,:]  (L2-resident re-read) ----------------
    {
        const int d    = tid & 63;
        const int part = tid >> 6;                 // 4 groups of 50 tokens
        const float* hp = hist + ((size_t)b * T_ + part * 50) * D_ + d;
        float accd = 0.f;
        #pragma unroll 5
        for (int i = 0; i < 50; i++)
            accd += sWgt[part * 50 + i] * hp[i * D_];
        sRed[tid] = accd;
    }
    __syncthreads();
    if (tid < D_)
        out[(size_t)b * D_ + tid] = sRed[tid] + sRed[64 + tid] + sRed[128 + tid] + sRed[192 + tid];
}

extern "C" void kernel_launch(void* const* d_in, const int* in_sizes, int n_in,
                              void* d_out, int out_size, void* d_ws, size_t ws_size,
                              hipStream_t stream) {
    const float* hist = (const float*)d_in[0];
    const float* targ = (const float*)d_in[1];
    const int*   mask = (const int*)d_in[2];
    const float* W1   = (const float*)d_in[3];
    const float* b1   = (const float*)d_in[4];
    const float* W2   = (const float*)d_in[5];
    const float* b2   = (const float*)d_in[6];
    const float* W3   = (const float*)d_in[7];
    const float* b3   = (const float*)d_in[8];
    float* out = (float*)d_out;

    const int B = in_sizes[1] / D_;   // target_emb is (B, 64)
    const size_t ws_needed = (size_t)(NB1 + NB2) * sizeof(_Float16);

    if (d_ws != nullptr && ws_size >= ws_needed) {
        _Float16* wf = (_Float16*)d_ws;
        prep_weights_kernel<<<dim3((NB1 + NB2 + 255) / 256), dim3(256), 0, stream>>>(W1, W2, wf);
        ta_mlp_attn_kernel<1><<<dim3(B), dim3(256), 0, stream>>>(
            hist, targ, mask, W1, b1, W2, b2, W3, b3, wf, out);
    } else {
        ta_mlp_attn_kernel<0><<<dim3(B), dim3(256), 0, stream>>>(
            hist, targ, mask, W1, b1, W2, b2, W3, b3, nullptr, out);
    }
    (void)n_in; (void)out_size;
}
